// LSTMTagger_18915035972398
// MI455X (gfx1250) — matrix operand order repaired and checked by reading, not compiled, not run on hardware
//
#include <hip/hip_runtime.h>
#include <hip/hip_bf16.h>
#include <math.h>

// Problem constants (match reference)
#define DD   512
#define LL   1024
#define KK   64
#define BB   16
#define NLL  16
#define SETD 128

typedef __attribute__((ext_vector_type(16))) __bf16 v16bf;
typedef __attribute__((ext_vector_type(8)))  float  v8f;

__device__ __forceinline__ v8f wmma_bf16(v16bf a, v16bf b, v8f c) {
  // D = A(16x32 bf16) * B(32x16 bf16) + C(16x16 f32)
  return __builtin_amdgcn_wmma_f32_16x16x32_bf16(false, a, false, b, (short)0, c,
                                                 false, false);
}

// A-fragment (16x32): source element (m,k) = A[m*lda + k] (row-major, f32 -> bf16)
// Layout: lanes 0-15 hold M=lane, K 0-15; lanes 16-31 hold M=lane-16, K 16-31
__device__ __forceinline__ v16bf load_a_rm(const float* A, int lda, int m0, int k0, int lane) {
  int m  = m0 + (lane & 15);
  int kb = (lane >> 4) << 4;                 // 0 or 16
  const float* p0 = A + (size_t)m * lda + k0 + kb;
  const float* p1 = p0 + 8;
  v16bf a;
#pragma unroll
  for (int i = 0; i < 8; ++i) { a[i] = (__bf16)p0[i]; a[8 + i] = (__bf16)p1[i]; }
  return a;
}

// B-fragment (32x16): source element (k,n) = Bs[k*ldb + n] (k-major, strided loads)
// Layout (contiguous k, same as load_a_rm): col = lane&15; lanes 0-15 hold K 0-15, lanes 16-31 hold K 16-31
__device__ __forceinline__ v16bf load_b_km(const float* Bs, int ldb, int k0, int n0, int lane) {
  int n  = n0 + (lane & 15);
  int kb = (lane >> 4) << 4;                 // 0 or 16
  const float* p = Bs + (size_t)(k0 + kb) * ldb + n;
  v16bf b;
#pragma unroll
  for (int j = 0; j < 16; ++j) b[j] = (__bf16)p[(size_t)j * ldb];
  return b;
}

// B-fragment where source element (k,n) = Bs[n*ldn + k] (contiguous in k)
__device__ __forceinline__ v16bf load_b_nm(const float* Bs, int ldn, int k0, int n0, int lane) {
  int n  = n0 + (lane & 15);
  int kb = (lane >> 4) << 4;
  const float* p = Bs + (size_t)n * ldn + k0 + kb;
  v16bf b;
#pragma unroll
  for (int j = 0; j < 16; ++j) b[j] = (__bf16)p[j];
  return b;
}

// ---------------------------------------------------------------------------
// A1: rate[b,l,k] = relu(emb[b,l,:] . W_att[layer,k,:] + b_att[layer,k])
//     stored TRANSPOSED: att_t[b, k, l].
// One wave computes a 16x64 macro-tile (all K heads): the emb A-fragment is
// loaded once per k-step and reused by 4 independent WMMAs (hides the XDL
// RAW hazard window and quarters the A traffic). grid (B*L/16) x 32 threads.
__global__ void k_rate_gemm(const float* __restrict__ emb, const float* __restrict__ W_att,
                            const float* __restrict__ b_att, float* __restrict__ att_t,
                            int layer) {
  int lane = threadIdx.x;
  int m0 = blockIdx.x << 4;                  // flattened row b*LL + l
  const float* Wa = W_att + (size_t)layer * KK * DD;   // [K, D] : B(k=d, n=head)=Wa[head*D+d]
  v8f acc0 = {}, acc1 = {}, acc2 = {}, acc3 = {};
  for (int k0 = 0; k0 < DD; k0 += 32) {
    __builtin_prefetch(emb + (size_t)(m0 + (lane & 15)) * DD + k0 + 32, 0, 1);
    v16bf a = load_a_rm(emb, DD, m0, k0, lane);
    v16bf b0 = load_b_nm(Wa, DD, k0,  0, lane);
    v16bf b1 = load_b_nm(Wa, DD, k0, 16, lane);
    v16bf b2 = load_b_nm(Wa, DD, k0, 32, lane);
    v16bf b3 = load_b_nm(Wa, DD, k0, 48, lane);
    acc0 = wmma_bf16(a, b0, acc0);
    acc1 = wmma_bf16(a, b1, acc1);
    acc2 = wmma_bf16(a, b2, acc2);
    acc3 = wmma_bf16(a, b3, acc3);
  }
  int bidx = m0 >> 10;                       // / LL
  int l0   = m0 & (LL - 1);
  int mrow = (lane >> 4) << 3;
  v8f accs[4] = {acc0, acc1, acc2, acc3};
#pragma unroll
  for (int nt = 0; nt < 4; ++nt) {
    int h    = (nt << 4) + (lane & 15);
    float bi = b_att[(size_t)layer * KK + h];
    float* dst = att_t + ((size_t)bidx * KK + h) * LL + l0 + mrow;
#pragma unroll
    for (int i = 0; i < 8; ++i) {
      float v = accs[nt][i] + bi;
      dst[i] = v > 0.f ? v : 0.f;
    }
  }
}

// ---------------------------------------------------------------------------
// Row softmax (length `rowlen`), in place. One 256-thread block per row.
__global__ void k_softmax_rows(float* __restrict__ data, int rowlen) {
  float* row = data + (size_t)blockIdx.x * rowlen;
  __shared__ float red[256];
  int t = threadIdx.x;
  float m = -3.4e38f;
  for (int i = t; i < rowlen; i += 256) m = fmaxf(m, row[i]);
  red[t] = m; __syncthreads();
  for (int s = 128; s > 0; s >>= 1) { if (t < s) red[t] = fmaxf(red[t], red[t + s]); __syncthreads(); }
  m = red[0]; __syncthreads();
  float sum = 0.f;
  for (int i = t; i < rowlen; i += 256) { float e = expf(row[i] - m); row[i] = e; sum += e; }
  red[t] = sum; __syncthreads();
  for (int s = 128; s > 0; s >>= 1) { if (t < s) red[t] += red[t + s]; __syncthreads(); }
  float inv = 1.f / red[0];
  __syncthreads();
  for (int i = t; i < rowlen; i += 256) row[i] *= inv;
}

// ---------------------------------------------------------------------------
// A3: sum_att[b,l] = sum_k att_t[b,k,l]; also writes att_dis[layer,b,l] into d_out.
__global__ void k_sumatt(const float* __restrict__ att_t, float* __restrict__ sum_att,
                         float* __restrict__ att_dis, int layer) {
  int idx = blockIdx.x * blockDim.x + threadIdx.x;    // b*LL + l, total B*L
  int b = idx >> 10, l = idx & (LL - 1);
  float s = 0.f;
#pragma unroll 4
  for (int k = 0; k < KK; ++k) s += att_t[((size_t)b * KK + k) * LL + l];
  sum_att[idx] = s;
  att_dis[(size_t)layer * BB * LL + idx] = s;
}

// ---------------------------------------------------------------------------
// B1: s_raw[b,k,d] = (1/sqrt(L)) * sum_l att_t[b,k,l] * emb[b,l,d]
// One wave computes a 64x16 macro-tile (all K heads x one d-tile): the
// EXPENSIVE strided emb B-fragment is loaded once per k-step and reused by 4
// independent WMMAs. grid (B, D/16) x 32 threads.
__global__ void k_pool_gemm(const float* __restrict__ att_t, const float* __restrict__ emb,
                            float* __restrict__ s_raw) {
  int lane = threadIdx.x;
  int b  = blockIdx.x;
  int n0 = blockIdx.y << 4;                  // d tile
  const float* A  = att_t + (size_t)b * KK * LL;     // [K, L]
  const float* Bm = emb   + (size_t)b * LL * DD;     // [L, D]
  v8f acc0 = {}, acc1 = {}, acc2 = {}, acc3 = {};
  for (int k0 = 0; k0 < LL; k0 += 32) {
    __builtin_prefetch(Bm + (size_t)(k0 + 32) * DD + n0, 0, 1);
    v16bf bb = load_b_km(Bm, DD, k0, n0, lane);      // strided: load once, use 4x
    v16bf a0 = load_a_rm(A, LL,  0, k0, lane);
    v16bf a1 = load_a_rm(A, LL, 16, k0, lane);
    v16bf a2 = load_a_rm(A, LL, 32, k0, lane);
    v16bf a3 = load_a_rm(A, LL, 48, k0, lane);
    acc0 = wmma_bf16(a0, bb, acc0);
    acc1 = wmma_bf16(a1, bb, acc1);
    acc2 = wmma_bf16(a2, bb, acc2);
    acc3 = wmma_bf16(a3, bb, acc3);
  }
  const float scale = 0.03125f;              // 1/sqrt(1024)
  int d    = n0 + (lane & 15);
  int mrow = (lane >> 4) << 3;
  v8f accs[4] = {acc0, acc1, acc2, acc3};
#pragma unroll
  for (int mt = 0; mt < 4; ++mt) {
    float* dst = s_raw + ((size_t)b * KK + (mt << 4) + mrow) * DD + d;
#pragma unroll
    for (int i = 0; i < 8; ++i) dst[(size_t)i * DD] = accs[mt][i] * scale;
  }
}

// ---------------------------------------------------------------------------
// B2: per (b,k): LN(s_raw; g_att,be_att) then vec_e = relu(x . W_pro[:,e] + b_pro),
//     scattered as out0[b, e*K + k]. One 256-thread block per (b,k) row.
__global__ void k_ln_pro(const float* __restrict__ s_raw, const float* __restrict__ g_att,
                         const float* __restrict__ be_att, const float* __restrict__ W_pro,
                         const float* __restrict__ b_pro, float* __restrict__ out0, int layer) {
  int bk = blockIdx.x;                       // b*KK + k
  int b = bk >> 6, k = bk & 63;
  const float* row = s_raw + (size_t)bk * DD;
  __shared__ float x[DD];
  __shared__ float red[256];
  int t = threadIdx.x;
  float x0 = row[t], x1 = row[t + 256];
  red[t] = x0 + x1; __syncthreads();
  for (int s = 128; s > 0; s >>= 1) { if (t < s) red[t] += red[t + s]; __syncthreads(); }
  float mean = red[0] * (1.f / DD); __syncthreads();
  float d0 = x0 - mean, d1 = x1 - mean;
  red[t] = d0 * d0 + d1 * d1; __syncthreads();
  for (int s = 128; s > 0; s >>= 1) { if (t < s) red[t] += red[t + s]; __syncthreads(); }
  float rstd = rsqrtf(red[0] * (1.f / DD) + 1e-5f); __syncthreads();
  const float* g  = g_att  + ((size_t)layer * KK + k) * DD;
  const float* be = be_att + ((size_t)layer * KK + k) * DD;
  x[t]       = d0 * rstd * g[t]       + be[t];
  x[t + 256] = d1 * rstd * g[t + 256] + be[t + 256];
  __syncthreads();
  // 8 outputs: thread t -> e = t&7, d-chunk c = t>>3 (32 chunks of 16)
  const float* Wp = W_pro + ((size_t)layer * KK + k) * DD * 8;
  int e = t & 7, c = t >> 3;
  float p = 0.f;
#pragma unroll
  for (int i = 0; i < 16; ++i) { int d = (c << 4) + i; p += x[d] * Wp[(size_t)d * 8 + e]; }
  red[t] = p; __syncthreads();
  for (int s = 128; s >= 8; s >>= 1) { if (t < s) red[t] += red[t + s]; __syncthreads(); }
  if (t < 8) {
    float v = red[t] + b_pro[((size_t)layer * KK + k) * 8 + t];
    out0[(size_t)b * DD + t * KK + k] = fmaxf(v, 0.f);
  }
}

// ---------------------------------------------------------------------------
// C: out1 = relu(out0 @ W_FI + b_FI) + out0 ; ott[b, layer*SET+:] = relu(out1 @ W_Set + b_Set)
// One block, 32 waves. M = 16 (== batch) fits one WMMA tile. out1 kept in LDS.
__global__ void k_fi_set(const float* __restrict__ out0, const float* __restrict__ W_FI,
                         const float* __restrict__ b_FI, const float* __restrict__ W_Set,
                         const float* __restrict__ b_Set, float* __restrict__ ott, int layer) {
  __shared__ float out1[BB * DD];            // 32 KB
  int lane = threadIdx.x & 31;
  int wid  = threadIdx.x >> 5;               // 0..31
  const float* Wf = W_FI + (size_t)layer * DD * DD;   // [Din, Dout]
  int n0 = wid << 4;
  v8f acc = {};
  for (int k0 = 0; k0 < DD; k0 += 32) {
    v16bf a  = load_a_rm(out0, DD, 0, k0, lane);
    v16bf bb = load_b_km(Wf, DD, k0, n0, lane);
    acc = wmma_bf16(a, bb, acc);
  }
  int n    = n0 + (lane & 15);
  int mrow = (lane >> 4) << 3;
  float bi = b_FI[(size_t)layer * DD + n];
#pragma unroll
  for (int i = 0; i < 8; ++i) {
    int m = mrow + i;
    float v = fmaxf(acc[i] + bi, 0.f) + out0[(size_t)m * DD + n];
    out1[m * DD + n] = v;
  }
  __syncthreads();
  if (wid < 8) {                             // SET = 128 -> 8 N-tiles
    const float* Ws = W_Set + (size_t)layer * DD * SETD;   // [D, SET]
    v8f acc2 = {};
    for (int k0 = 0; k0 < DD; k0 += 32) {
      v16bf a  = load_a_rm((const float*)out1, DD, 0, k0, lane);  // LDS reads
      v16bf bb = load_b_km(Ws, SETD, k0, n0, lane);
      acc2 = wmma_bf16(a, bb, acc2);
    }
    int n2   = n0 + (lane & 15);
    float b2 = b_Set[(size_t)layer * SETD + n2];
#pragma unroll
    for (int i = 0; i < 8; ++i) {
      int m = mrow + i;
      ott[(size_t)m * (NLL * SETD) + layer * SETD + n2] = fmaxf(acc2[i] + b2, 0.f);
    }
  }
}

// ---------------------------------------------------------------------------
// D: emb[b,l,:] = LN(emb*(sum_att*wp1 + bp1) + emb ; g_n, be_n), in place (row-local).
__global__ void k_update_emb(float* __restrict__ emb, const float* __restrict__ sum_att,
                             const float* __restrict__ w_p1, const float* __restrict__ b_p1,
                             const float* __restrict__ g_n, const float* __restrict__ be_n,
                             int layer) {
  int r = blockIdx.x;                        // b*LL + l
  int t = threadIdx.x;
  float sa = sum_att[r];
  float* row = emb + (size_t)r * DD;
  const float* wp = w_p1 + (size_t)layer * DD;
  const float* bp = b_p1 + (size_t)layer * DD;
  const float* g  = g_n  + (size_t)layer * DD;
  const float* be = be_n + (size_t)layer * DD;
  __shared__ float red[256];
  float x0 = row[t], x1 = row[t + 256];
  float y0 = fmaf(x0, fmaf(sa, wp[t], bp[t]), x0);
  float y1 = fmaf(x1, fmaf(sa, wp[t + 256], bp[t + 256]), x1);
  red[t] = y0 + y1; __syncthreads();
  for (int s = 128; s > 0; s >>= 1) { if (t < s) red[t] += red[t + s]; __syncthreads(); }
  float mean = red[0] * (1.f / DD); __syncthreads();
  float d0 = y0 - mean, d1 = y1 - mean;
  red[t] = d0 * d0 + d1 * d1; __syncthreads();
  for (int s = 128; s > 0; s >>= 1) { if (t < s) red[t] += red[t + s]; __syncthreads(); }
  float rstd = rsqrtf(red[0] * (1.f / DD) + 1e-5f);
  row[t]       = d0 * rstd * g[t]       + be[t];
  row[t + 256] = d1 * rstd * g[t + 256] + be[t + 256];
}

// ---------------------------------------------------------------------------
// Final AAt: rate_f[b,l] = relu(emb[b,l,:].w_aat + b_aat). One wave per row.
__global__ void k_aat_rate(const float* __restrict__ emb, const float* __restrict__ w_aat,
                           const float* __restrict__ b_aat, float* __restrict__ rate_f) {
  int lane = threadIdx.x & 31, wid = threadIdx.x >> 5;
  int r = blockIdx.x * 8 + wid;
  const float* row = emb + (size_t)r * DD;
  float p = 0.f;
  for (int i = lane; i < DD; i += 32) p += row[i] * w_aat[i];
#pragma unroll
  for (int off = 16; off > 0; off >>= 1) p += __shfl_down(p, off, 32);
  if (lane == 0) rate_f[r] = fmaxf(p + b_aat[0], 0.f);
}

// spool[b,d] = (1/sqrt(L)) * sum_l att_f[b,l]*emb[b,l,d]
__global__ void k_aat_pool(const float* __restrict__ emb, const float* __restrict__ att_f,
                           float* __restrict__ spool) {
  int idx = blockIdx.x * blockDim.x + threadIdx.x;   // b*DD + d
  int b = idx >> 9, d = idx & (DD - 1);
  const float* e = emb + (size_t)b * LL * DD + d;
  const float* a = att_f + (size_t)b * LL;
  float s = 0.f;
  for (int l = 0; l < LL; ++l) s += a[l] * e[(size_t)l * DD];
  spool[idx] = s * 0.03125f;
}

// In-place LN of spool rows (len 512) with g_aat/be_aat.
__global__ void k_ln_final(float* __restrict__ spool, const float* __restrict__ g,
                           const float* __restrict__ be) {
  float* row = spool + (size_t)blockIdx.x * DD;
  __shared__ float red[256];
  int t = threadIdx.x;
  float x0 = row[t], x1 = row[t + 256];
  red[t] = x0 + x1; __syncthreads();
  for (int s = 128; s > 0; s >>= 1) { if (t < s) red[t] += red[t + s]; __syncthreads(); }
  float mean = red[0] * (1.f / DD); __syncthreads();
  float d0 = x0 - mean, d1 = x1 - mean;
  red[t] = d0 * d0 + d1 * d1; __syncthreads();
  for (int s = 128; s > 0; s >>= 1) { if (t < s) red[t] += red[t + s]; __syncthreads(); }
  float rstd = rsqrtf(red[0] * (1.f / DD) + 1e-5f);
  row[t]       = d0 * rstd * g[t]       + be[t];
  row[t + 256] = d1 * rstd * g[t + 256] + be[t + 256];
}

// P[b] = sigmoid([spool_ln | ott] . W_h2p + b_h2p)
__global__ void k_head(const float* __restrict__ spool, const float* __restrict__ ott,
                       const float* __restrict__ W_h2p, const float* __restrict__ b_h2p,
                       float* __restrict__ P) {
  int b = blockIdx.x, t = threadIdx.x;
  __shared__ float red[256];
  float p = 0.f;
  for (int i = t; i < DD; i += 256) p += spool[(size_t)b * DD + i] * W_h2p[i];
  for (int j = t; j < NLL * SETD; j += 256)
    p += ott[(size_t)b * (NLL * SETD) + j] * W_h2p[DD + j];
  red[t] = p; __syncthreads();
  for (int s = 128; s > 0; s >>= 1) { if (t < s) red[t] += red[t + s]; __syncthreads(); }
  if (t == 0) P[b] = 1.f / (1.f + expf(-(red[0] + b_h2p[0])));
}

// ---------------------------------------------------------------------------
extern "C" void kernel_launch(void* const* d_in, const int* in_sizes, int n_in,
                              void* d_out, int out_size, void* d_ws, size_t ws_size,
                              hipStream_t stream) {
  const float* embding = (const float*)d_in[0];
  const float* W_att   = (const float*)d_in[1];
  const float* b_att   = (const float*)d_in[2];
  const float* g_att   = (const float*)d_in[3];
  const float* be_att  = (const float*)d_in[4];
  const float* W_pro   = (const float*)d_in[5];
  const float* b_pro   = (const float*)d_in[6];
  const float* w_p1    = (const float*)d_in[7];
  const float* b_p1    = (const float*)d_in[8];
  const float* W_FI    = (const float*)d_in[9];
  const float* b_FI    = (const float*)d_in[10];
  const float* W_Set   = (const float*)d_in[11];
  const float* b_Set   = (const float*)d_in[12];
  const float* g_n     = (const float*)d_in[13];
  const float* be_n    = (const float*)d_in[14];
  const float* w_aat   = (const float*)d_in[15];
  const float* b_aat   = (const float*)d_in[16];
  const float* g_aat   = (const float*)d_in[17];
  const float* be_aat  = (const float*)d_in[18];
  const float* W_h2p   = (const float*)d_in[19];
  const float* b_h2p   = (const float*)d_in[20];

  float* out = (float*)d_out;                // [0..15] = P ; [16..] = att_dis

  // Workspace carve-up (floats)
  const size_t N_EMB  = (size_t)BB * LL * DD;        // 8,388,608
  const size_t N_ATT  = (size_t)BB * KK * LL;        // 1,048,576
  const size_t N_SATT = (size_t)BB * LL;             // 16,384
  const size_t N_SRAW = (size_t)BB * KK * DD;        // 524,288
  const size_t N_OUT0 = (size_t)BB * DD;             // 8,192
  const size_t N_OTT  = (size_t)BB * NLL * SETD;     // 32,768
  const size_t needed = N_EMB + N_ATT + N_SATT + N_SRAW + N_OUT0 + N_OTT +
                        N_SATT /*rate_f*/ + N_OUT0 /*spool*/;
  if (ws_size < needed * sizeof(float)) return;      // deterministic guard

  float* ws      = (float*)d_ws;
  float* emb     = ws;
  float* att_t   = emb + N_EMB;
  float* sum_att = att_t + N_ATT;
  float* s_raw   = sum_att + N_SATT;
  float* out0    = s_raw + N_SRAW;
  float* ott     = out0 + N_OUT0;
  float* rate_f  = ott + N_OTT;
  float* spool   = rate_f + N_SATT;

  // State starts as the input embedding; updated in place each layer.
  hipMemcpyAsync(emb, embding, N_EMB * sizeof(float), hipMemcpyDeviceToDevice, stream);

  for (int layer = 0; layer < NLL; ++layer) {
    k_rate_gemm <<<(BB * LL) / 16, 32, 0, stream>>>(emb, W_att, b_att, att_t, layer);
    k_softmax_rows<<<BB * KK, 256, 0, stream>>>(att_t, LL);
    k_sumatt    <<<(BB * LL) / 256, 256, 0, stream>>>(att_t, sum_att, out + BB, layer);
    k_pool_gemm <<<dim3(BB, DD / 16), 32, 0, stream>>>(att_t, emb, s_raw);
    k_ln_pro    <<<BB * KK, 256, 0, stream>>>(s_raw, g_att, be_att, W_pro, b_pro, out0, layer);
    k_fi_set    <<<1, 1024, 0, stream>>>(out0, W_FI, b_FI, W_Set, b_Set, ott, layer);
    k_update_emb<<<BB * LL, 256, 0, stream>>>(emb, sum_att, w_p1, b_p1, g_n, be_n, layer);
  }

  k_aat_rate    <<<(BB * LL) / 8, 256, 0, stream>>>(emb, w_aat, b_aat, rate_f);
  k_softmax_rows<<<BB, 256, 0, stream>>>(rate_f, LL);
  k_aat_pool    <<<(BB * DD) / 256, 256, 0, stream>>>(emb, rate_f, spool);
  k_ln_final    <<<BB, 256, 0, stream>>>(spool, g_aat, be_aat);
  k_head        <<<BB, 256, 0, stream>>>(spool, ott, W_h2p, b_h2p, out);
}
